// ESBN_51848845197901
// MI455X (gfx1250) — matrix operand order repaired and checked by reading, not compiled, not run on hardware
//
#include <hip/hip_runtime.h>
#include <hip/hip_bf16.h>

// ---------------- problem constants ----------------
#define BATCH 512
#define TT    32
#define NSLOT 33      // T + 1
#define DDIM  1024    // 32*32
#define ZDIM  128
#define KDIM  256
#define PDIM  257     // K + 1
#define HID   512
#define G4    2048    // 4*HID
#define YD    4

// K-tile counts (K padded to multiple of 32)
#define KT_ENC  32    // 1024/32
#define KT_W    9     // ceil(257/32)
#define KT_U    16    // 512/32
#define KT_KEY  16    // 512/32
#define FRAG    512   // halves per packed 16x32 (or 32x16) tile: 32 lanes * 16

typedef __attribute__((ext_vector_type(16))) _Float16 v16h;
typedef __attribute__((ext_vector_type(8)))  float    v8f;

__device__ __forceinline__ float sigmoidf(float x) { return 1.0f / (1.0f + expf(-x)); }

#define WMMA_F16(A, B, Cacc) \
  __builtin_amdgcn_wmma_f32_16x16x32_f16(false, (A), false, (B), (short)0, (Cacc), false, false)

// ---------------- packing: f32 row-major -> f16 WMMA fragment layout ----------------
// A 16x32 (MxK) 16-bit layout: lanes 0-15 m=lane; K = {kb..kb+7, 16+kb..16+kb+7}, kb=(lane>>4)*8
// one thread per (tile, lane); writes 16 contiguous halves (32B).
__global__ __launch_bounds__(256)
void pack_a_kernel(const float* __restrict__ A, int lda, int M, int K, int KT,
                   _Float16* __restrict__ out) {
  const int tid  = blockIdx.x * 256 + threadIdx.x;
  const int lane = tid & 31;
  const int tile = tid >> 5;
  if (tile >= (M >> 4) * KT) return;
  const int i  = tile / KT;           // m tile
  const int j  = tile % KT;           // k tile
  const int m  = lane & 15;
  const int kb = (lane >> 4) * 8;
  const float* Ar = A + (size_t)(i * 16 + m) * lda + j * 32;
  v16h f;
#pragma unroll
  for (int e = 0; e < 16; ++e) {
    const int k = (e < 8) ? (kb + e) : (16 + kb + (e - 8));
    f[e] = (j * 32 + k < K) ? (_Float16)Ar[k] : (_Float16)0.0f;
  }
  *(v16h*)(out + (size_t)tile * FRAG + lane * 16) = f;
}

// B 32x16 (KxN) 16-bit layout: n=lane&15; K = {kb..kb+7, 16+kb..16+kb+7}, kb=(lane>>4)*8
// tiles ordered (ntile, ktile) so the GEMM walks K contiguously.
__global__ __launch_bounds__(256)
void pack_b_kernel(const float* __restrict__ Bm, int N, int K, int KT,
                   _Float16* __restrict__ out) {
  const int tid  = blockIdx.x * 256 + threadIdx.x;
  const int lane = tid & 31;
  const int tile = tid >> 5;
  if (tile >= (N >> 4) * KT) return;
  const int nt = tile / KT;
  const int j  = tile % KT;
  const int n  = nt * 16 + (lane & 15);
  const int kb = (lane >> 4) * 8;
  v16h f;
#pragma unroll
  for (int e = 0; e < 16; ++e) {
    const int k = j * 32 + ((e < 8) ? (kb + e) : (16 + kb + (e - 8)));
    f[e] = (k < K) ? (_Float16)Bm[(size_t)k * N + n] : (_Float16)0.0f;
  }
  *(v16h*)(out + (size_t)tile * FRAG + lane * 16) = f;
}

// ---------------- WMMA GEMM on pre-packed fragments, 2x2 register blocking ----------------
// C(MxN) = act( A1*B1 [+ A2*B2] + bias ); packed A: (mtile, ktile), packed B: (ntile, ktile).
// each wave computes a 32x32 output block (4 wmma per k-step, each fragment reused twice).
// grid: (N/32, M/128), block 128 (4 waves).
template <int KT1, int KT2>
__global__ __launch_bounds__(128)
void gemm_packed(const _Float16* __restrict__ Ap, const _Float16* __restrict__ Bp,
                 const _Float16* __restrict__ Ap2, const _Float16* __restrict__ Bp2,
                 const float* __restrict__ bias, float* __restrict__ C, int N, int act) {
  const int lane = threadIdx.x & 31;
  const int wave = threadIdx.x >> 5;
  const int mt0  = (blockIdx.y * 4 + wave) * 2;   // first of 2 m-tiles
  const int nt0  = blockIdx.x * 2;                // first of 2 n-tiles

  v8f acc00 = {0.f, 0.f, 0.f, 0.f, 0.f, 0.f, 0.f, 0.f};
  v8f acc01 = acc00, acc10 = acc00, acc11 = acc00;

  {
    const _Float16* a0 = Ap + ((size_t)mt0 * KT1) * FRAG + lane * 16;
    const _Float16* a1 = a0 + (size_t)KT1 * FRAG;
    const _Float16* b0 = Bp + ((size_t)nt0 * KT1) * FRAG + lane * 16;
    const _Float16* b1 = b0 + (size_t)KT1 * FRAG;
#pragma unroll
    for (int j = 0; j < KT1; ++j) {
      const v16h A0 = *(const v16h*)(a0 + (size_t)j * FRAG);
      const v16h A1 = *(const v16h*)(a1 + (size_t)j * FRAG);
      const v16h B0 = *(const v16h*)(b0 + (size_t)j * FRAG);
      const v16h B1 = *(const v16h*)(b1 + (size_t)j * FRAG);
      acc00 = WMMA_F16(A0, B0, acc00);
      acc01 = WMMA_F16(A0, B1, acc01);
      acc10 = WMMA_F16(A1, B0, acc10);
      acc11 = WMMA_F16(A1, B1, acc11);
    }
  }
  if constexpr (KT2 > 0) {
    const _Float16* a0 = Ap2 + ((size_t)mt0 * KT2) * FRAG + lane * 16;
    const _Float16* a1 = a0 + (size_t)KT2 * FRAG;
    const _Float16* b0 = Bp2 + ((size_t)nt0 * KT2) * FRAG + lane * 16;
    const _Float16* b1 = b0 + (size_t)KT2 * FRAG;
#pragma unroll
    for (int j = 0; j < KT2; ++j) {
      const v16h A0 = *(const v16h*)(a0 + (size_t)j * FRAG);
      const v16h A1 = *(const v16h*)(a1 + (size_t)j * FRAG);
      const v16h B0 = *(const v16h*)(b0 + (size_t)j * FRAG);
      const v16h B1 = *(const v16h*)(b1 + (size_t)j * FRAG);
      acc00 = WMMA_F16(A0, B0, acc00);
      acc01 = WMMA_F16(A0, B1, acc01);
      acc10 = WMMA_F16(A1, B0, acc10);
      acc11 = WMMA_F16(A1, B1, acc11);
    }
  }

  // C/D layout: lanes 0-15 -> n=lane, M=row0+r ; lanes 16-31 -> n=lane-16, M=row0+8+r
  const int nlo  = lane & 15;
  const int rsub = (lane >> 4) << 3;
#pragma unroll
  for (int ti = 0; ti < 2; ++ti) {
#pragma unroll
    for (int tj = 0; tj < 2; ++tj) {
      const v8f acc = (ti == 0) ? ((tj == 0) ? acc00 : acc01)
                                : ((tj == 0) ? acc10 : acc11);
      const int n     = (nt0 + tj) * 16 + nlo;
      const int rbase = (mt0 + ti) * 16 + rsub;
      const float bv  = bias ? bias[n] : 0.0f;
#pragma unroll
      for (int r = 0; r < 8; ++r) {
        float v = acc[r] + bv;
        if (act == 1) v = fmaxf(v, 0.0f);
        C[(size_t)(rbase + r) * N + n] = v;
      }
    }
  }
}

// ---------------- zero init ----------------
__global__ void zero_kernel(float* p, size_t n) {
  size_t i = (size_t)blockIdx.x * blockDim.x + threadIdx.x;
  const size_t stride = (size_t)gridDim.x * blockDim.x;
  for (; i < n; i += stride) p[i] = 0.0f;
}

// ---------------- batch-axis norm: z(B,T,Z) -> zfull(B,N,Z) slots 0..T-1 ----------------
__global__ __launch_bounds__(256)
void bn_kernel(const float* __restrict__ z, const float* __restrict__ gamma,
               const float* __restrict__ beta, float* __restrict__ zfull) {
  const int col = blockIdx.x;          // t*ZDIM + zi
  const int t  = col / ZDIM;
  const int zi = col % ZDIM;
  __shared__ float s1[256], s2[256];
  float a = 0.f, b = 0.f;
  for (int bb = threadIdx.x; bb < BATCH; bb += 256) {
    const float v = z[(size_t)bb * TT * ZDIM + col];
    a += v; b += v * v;
  }
  s1[threadIdx.x] = a; s2[threadIdx.x] = b;
  __syncthreads();
  for (int s = 128; s > 0; s >>= 1) {
    if (threadIdx.x < s) { s1[threadIdx.x] += s1[threadIdx.x + s]; s2[threadIdx.x] += s2[threadIdx.x + s]; }
    __syncthreads();
  }
  const float mu  = s1[0] * (1.0f / BATCH);
  const float var = s2[0] * (1.0f / BATCH) - mu * mu;
  const float rs  = rsqrtf(var + 1e-8f);
  const float gm = gamma[zi], bt = beta[zi];
  for (int bb = threadIdx.x; bb < BATCH; bb += 256) {
    const float v = z[(size_t)bb * TT * ZDIM + col];
    zfull[(size_t)bb * NSLOT * ZDIM + (size_t)t * ZDIM + zi] = (v - mu) * rs * gm + bt;
  }
}

// ---------------- LSTM cell (inputs swapped per the faithful reference) ----------------
__global__ __launch_bounds__(256)
void cell_kernel(const float* __restrict__ gates, const float* __restrict__ h_old,
                 float* __restrict__ h_new, float* __restrict__ c_new) {
  const int idx = blockIdx.x * 256 + threadIdx.x;      // B*HID
  if (idx >= BATCH * HID) return;
  const int b = idx / HID, j = idx % HID;
  const float* gr = gates + (size_t)b * G4;
  const float ig = sigmoidf(gr[j]);
  const float fg = sigmoidf(gr[HID + j]);
  const float gg = tanhf(gr[2 * HID + j]);
  const float og = sigmoidf(gr[3 * HID + j]);
  const float cn = fg * h_old[idx] + ig * gg;
  c_new[idx] = cn;
  h_new[idx] = og * tanhf(cn);
}

// ---------------- gate g and (final-step) y heads ----------------
__global__ __launch_bounds__(256)
void head_kernel(const float* __restrict__ h, const float* __restrict__ Wg,
                 const float* __restrict__ bg, const float* __restrict__ Wy,
                 const float* __restrict__ by, float* __restrict__ gval,
                 float* __restrict__ yout, int final_step) {
  const int b = blockIdx.x;
  const int tid = threadIdx.x;
  __shared__ float red[5][256];
  float ag = 0.f, a0 = 0.f, a1 = 0.f, a2 = 0.f, a3 = 0.f;
  for (int i = tid; i < HID; i += 256) {
    const float hv = h[(size_t)b * HID + i];
    ag += hv * Wg[i];
    a0 += hv * Wy[i * YD + 0];
    a1 += hv * Wy[i * YD + 1];
    a2 += hv * Wy[i * YD + 2];
    a3 += hv * Wy[i * YD + 3];
  }
  red[0][tid] = ag; red[1][tid] = a0; red[2][tid] = a1; red[3][tid] = a2; red[4][tid] = a3;
  __syncthreads();
  for (int s = 128; s > 0; s >>= 1) {
    if (tid < s)
      for (int q = 0; q < 5; ++q) red[q][tid] += red[q][tid + s];
    __syncthreads();
  }
  if (tid == 0) {
    gval[b] = sigmoidf(red[0][0] + bg[0]);
    if (final_step) {
      float y[YD];
      float mx = -1e30f;
      for (int j = 0; j < YD; ++j) { y[j] = red[1 + j][0] + by[j]; mx = fmaxf(mx, y[j]); }
      float s = 0.f;
      for (int j = 0; j < YD; ++j) { y[j] = expf(y[j] - mx); s += y[j]; }
      const float inv = 1.0f / s;
      for (int j = 0; j < YD; ++j) yout[b * YD + j] = y[j] * inv;
    }
  }
}

// ---------------- content-addressed memory step ----------------
__global__ __launch_bounds__(256)
void memory_kernel(const float* __restrict__ zfull, float* __restrict__ M_k,
                   float* __restrict__ M_v, const float* __restrict__ key_w,
                   const float* __restrict__ gval, float* __restrict__ key_r,
                   const float* __restrict__ cgain, const float* __restrict__ cbias,
                   int t) {
  const int b = blockIdx.x;
  const int tid = threadIdx.x;
  __shared__ float zt[ZDIM];
  __shared__ float simS[NSLOT], wk[NSLOT], ck[NSLOT];

  const float* zrow = zfull + ((size_t)b * NSLOT + t) * ZDIM;
  if (tid < ZDIM) zt[tid] = zrow[tid];
  __syncthreads();

  if (tid < NSLOT) {
    const float* mv = M_v + ((size_t)b * NSLOT + tid) * ZDIM;
    float s = 0.f;
    for (int zi = 0; zi < ZDIM; ++zi) s += zt[zi] * mv[zi];
    simS[tid] = s;
    ck[tid] = sigmoidf(s * cgain[0] + cbias[0]);
  }
  __syncthreads();

  if (tid == 0) {
    if (t > 0) {
      float mx = -1e30f;
      for (int n = 0; n < t; ++n) mx = fmaxf(mx, simS[n]);
      float sum = 0.f;
      for (int n = 0; n < NSLOT; ++n) {
        const float e = (n < t) ? expf(simS[n] - mx) : 0.0f;
        wk[n] = e; sum += e;
      }
      const float inv = 1.0f / sum;
      for (int n = 0; n < NSLOT; ++n) wk[n] *= inv;
    } else {
      for (int n = 0; n < NSLOT; ++n) wk[n] = 0.0f;
    }
  }
  __syncthreads();

  const float g = gval[b];
  for (int p = tid; p < PDIM; p += 256) {
    float w = 0.f;
    for (int n = 0; n < NSLOT; ++n) {
      const float m = (p < KDIM) ? M_k[((size_t)b * NSLOT + n) * KDIM + p] : ck[n];
      w += wk[n] * m;
    }
    key_r[(size_t)b * PDIM + p] = (t == 0) ? 0.0f : g * w;
  }
  __syncthreads();

  // write slot t AFTER all reads of the pre-step memory
  for (int p = tid; p < KDIM; p += 256)
    M_k[((size_t)b * NSLOT + t) * KDIM + p] = key_w[(size_t)b * KDIM + p];
  for (int zi = tid; zi < ZDIM; zi += 256)
    M_v[((size_t)b * NSLOT + t) * ZDIM + zi] = zt[zi];
}

// ---------------- host orchestration ----------------
extern "C" void kernel_launch(void* const* d_in, const int* in_sizes, int n_in,
                              void* d_out, int out_size, void* d_ws, size_t ws_size,
                              hipStream_t stream) {
  (void)in_sizes; (void)n_in; (void)out_size; (void)ws_size;

  const float* x_seq  = (const float*)d_in[0];
  const float* W_enc  = (const float*)d_in[1];
  const float* b_enc  = (const float*)d_in[2];
  const float* W_lstm = (const float*)d_in[3];
  const float* U_lstm = (const float*)d_in[4];
  const float* b_lstm = (const float*)d_in[5];
  const float* W_key  = (const float*)d_in[6];
  const float* b_key  = (const float*)d_in[7];
  const float* W_g    = (const float*)d_in[8];
  const float* b_g    = (const float*)d_in[9];
  const float* W_y    = (const float*)d_in[10];
  const float* b_y    = (const float*)d_in[11];
  const float* gamma  = (const float*)d_in[12];
  const float* beta   = (const float*)d_in[13];
  const float* cgain  = (const float*)d_in[14];
  const float* cbias  = (const float*)d_in[15];
  float* yout = (float*)d_out;

  // --- f32 workspace region ---
  float* ws = (float*)d_ws;
  size_t off = 0;
  float* zfull = ws + off; off += (size_t)BATCH * NSLOT * ZDIM;
  float* h0    = ws + off; off += (size_t)BATCH * HID;
  float* h1    = ws + off; off += (size_t)BATCH * HID;
  float* c0    = ws + off; off += (size_t)BATCH * HID;
  float* c1    = ws + off; off += (size_t)BATCH * HID;
  float* key_r = ws + off; off += (size_t)BATCH * PDIM;   // 131,584
  off += 128 - (off & 127);                                // realign
  float* M_k   = ws + off; off += (size_t)BATCH * NSLOT * KDIM;
  float* M_v   = ws + off; off += (size_t)BATCH * NSLOT * ZDIM;
  const size_t zero_span = off;                            // state region to zero
  float* zbuf  = ws + off; off += (size_t)BATCH * TT * ZDIM;
  float* gates = ws + off; off += (size_t)BATCH * G4;
  float* key_w = ws + off; off += (size_t)BATCH * KDIM;
  float* gval  = ws + off; off += (size_t)BATCH;
  off += 128 - (off & 127);

  // --- f16 packed-fragment region (32B-aligned; all sizes multiples of FRAG) ---
  _Float16* hb = (_Float16*)(ws + off);
  size_t ho = 0;
  _Float16* xp     = hb + ho; ho += (size_t)(BATCH * TT / 16) * KT_ENC * FRAG; // 16.8M
  _Float16* Wencp  = hb + ho; ho += (size_t)(ZDIM / 16) * KT_ENC * FRAG;
  _Float16* Wlstmp = hb + ho; ho += (size_t)(G4 / 16) * KT_W * FRAG;
  _Float16* Ulstmp = hb + ho; ho += (size_t)(G4 / 16) * KT_U * FRAG;
  _Float16* Wkeyp  = hb + ho; ho += (size_t)(KDIM / 16) * KT_KEY * FRAG;
  _Float16* keyrp  = hb + ho; ho += (size_t)(BATCH / 16) * KT_W * FRAG;
  _Float16* cp     = hb + ho; ho += (size_t)(BATCH / 16) * KT_U * FRAG;
  _Float16* hp     = hb + ho; ho += (size_t)(BATCH / 16) * KT_KEY * FRAG;

  // 1) zero initial state + memory + zfull (null slot included)
  zero_kernel<<<2048, 256, 0, stream>>>(ws, zero_span);

  // 2) pack constants (once per launch) and x
  pack_a_kernel<<<((BATCH * TT / 16) * KT_ENC * 32) / 256, 256, 0, stream>>>(
      x_seq, DDIM, BATCH * TT, DDIM, KT_ENC, xp);
  pack_b_kernel<<<((ZDIM / 16) * KT_ENC * 32) / 256, 256, 0, stream>>>(
      W_enc, ZDIM, DDIM, KT_ENC, Wencp);
  pack_b_kernel<<<((G4 / 16) * KT_W * 32) / 256, 256, 0, stream>>>(
      W_lstm, G4, PDIM, KT_W, Wlstmp);
  pack_b_kernel<<<((G4 / 16) * KT_U * 32) / 256, 256, 0, stream>>>(
      U_lstm, G4, HID, KT_U, Ulstmp);
  pack_b_kernel<<<((KDIM / 16) * KT_KEY * 32) / 256, 256, 0, stream>>>(
      W_key, KDIM, HID, KT_KEY, Wkeyp);

  // 3) encoder: z = relu(x @ W_enc + b_enc)   (16384 x 128, K=1024)
  gemm_packed<KT_ENC, 0><<<dim3(ZDIM / 32, (BATCH * TT) / 128), 128, 0, stream>>>(
      xp, Wencp, nullptr, nullptr, b_enc, zbuf, ZDIM, 1);

  // 4) batch-axis norm -> zfull slots 0..T-1 (slot T stays zero)
  bn_kernel<<<TT * ZDIM, 256, 0, stream>>>(zbuf, gamma, beta, zfull);

  // 5) recurrent scan, 33 steps
  float* hA = h0; float* hB = h1;
  float* cA = c0; float* cB = c1;
  for (int t = 0; t < NSLOT; ++t) {
    // pack activations for the gate GEMM
    pack_a_kernel<<<((BATCH / 16) * KT_W * 32) / 256, 256, 0, stream>>>(
        key_r, PDIM, BATCH, PDIM, KT_W, keyrp);
    pack_a_kernel<<<((BATCH / 16) * KT_U * 32) / 256, 256, 0, stream>>>(
        cA, HID, BATCH, HID, KT_U, cp);

    // gates = key_r @ W_lstm + c_old @ U_lstm + b_lstm   (512 x 2048)
    gemm_packed<KT_W, KT_U><<<dim3(G4 / 32, BATCH / 128), 128, 0, stream>>>(
        keyrp, Wlstmp, cp, Ulstmp, b_lstm, gates, G4, 0);

    // cell (c_in = h_old per the swapped reference)
    cell_kernel<<<(BATCH * HID) / 256, 256, 0, stream>>>(gates, hA, hB, cB);

    // key_w = relu(h_new @ W_key + b_key)   (512 x 256)
    pack_a_kernel<<<((BATCH / 16) * KT_KEY * 32) / 256, 256, 0, stream>>>(
        hB, HID, BATCH, HID, KT_KEY, hp);
    gemm_packed<KT_KEY, 0><<<dim3(KDIM / 32, BATCH / 128), 128, 0, stream>>>(
        hp, Wkeyp, nullptr, nullptr, b_key, key_w, KDIM, 1);

    // g and (final-step) y heads
    head_kernel<<<BATCH, 256, 0, stream>>>(hB, W_g, b_g, W_y, b_y, gval, yout,
                                           (t == NSLOT - 1) ? 1 : 0);

    // memory read/update -> key_r, write slot t
    memory_kernel<<<BATCH, 256, 0, stream>>>(zfull, M_k, M_v, key_w, gval, key_r,
                                             cgain, cbias, t);

    // ping-pong carries
    float* tmp;
    tmp = hA; hA = hB; hB = tmp;
    tmp = cA; cA = cB; cB = tmp;
  }
}
